// Head_80951543595566
// MI455X (gfx1250) — matrix operand order repaired and checked by reading, not compiled, not run on hardware
//
#include <hip/hip_runtime.h>

// Problem constants (fixed by the reference)
#define BB 16
#define TT 2048
#define CC 1024
#define HH 64
#define NTILE 128          // TT/16 row tiles per batch

typedef __attribute__((ext_vector_type(16))) __bf16 bf16x16;
typedef __attribute__((ext_vector_type(8)))  float  f32x8;

union BFrag {
  bf16x16 f;
  uint4 u[2];
  unsigned short h[16];
};

__device__ __forceinline__ unsigned short f2bf_bits(float x) {
  unsigned int u = __builtin_bit_cast(unsigned int, x);
  unsigned int r = u + 0x7FFFu + ((u >> 16) & 1u);   // round-to-nearest-even
  return (unsigned short)(r >> 16);
}

__device__ __forceinline__ float rowmax16(float v) {
  v = fmaxf(v, __shfl_xor(v, 1, 32));
  v = fmaxf(v, __shfl_xor(v, 2, 32));
  v = fmaxf(v, __shfl_xor(v, 4, 32));
  v = fmaxf(v, __shfl_xor(v, 8, 32));
  return v;
}
__device__ __forceinline__ float rowsum16(float v) {
  v += __shfl_xor(v, 1, 32);
  v += __shfl_xor(v, 2, 32);
  v += __shfl_xor(v, 4, 32);
  v += __shfl_xor(v, 8, 32);
  return v;
}

// ---------------------------------------------------------------------------
// Kernel 0: repack Wq/Wk/Wv (fp32 [C,H]) into bf16 WMMA B-fragment layout.
// Entry (mat, chunk, n, lane): h[j] = W[chunk*32 + 16*(lane>>4) + j][n*16 + lane&15]
// ---------------------------------------------------------------------------
__global__ void __launch_bounds__(32)
wpack_kernel(const float* __restrict__ Wq, const float* __restrict__ Wk,
             const float* __restrict__ Wv, unsigned short* __restrict__ wpack) {
  int gid  = blockIdx.x * 32 + threadIdx.x;   // 12288 entries
  int lane = gid & 31;
  int n    = (gid >> 5) & 3;
  int ch   = (gid >> 7) & 31;
  int mat  = gid >> 12;                       // 0..2
  const float* W = (mat == 0) ? Wq : ((mat == 1) ? Wk : Wv);
  int c0   = ch * 32 + ((lane >> 4) << 4);
  int hcol = n * 16 + (lane & 15);
  unsigned int w[8];
#pragma unroll
  for (int p = 0; p < 8; ++p) {
    unsigned int lo = f2bf_bits(W[(size_t)(c0 + 2 * p    ) * HH + hcol]);
    unsigned int hi = f2bf_bits(W[(size_t)(c0 + 2 * p + 1) * HH + hcol]);
    w[p] = lo | (hi << 16);
  }
  unsigned short* dst = wpack + (size_t)gid * 16;
  *(uint4*)dst       = *(uint4*)&w[0];
  *(uint4*)(dst + 8) = *(uint4*)&w[4];
}

// ---------------------------------------------------------------------------
// Async staging of one 32-wide K-chunk of the packed weights into LDS.
// 12 fragment blocks x 1KB; each lane copies 32B per block via 2x async b128.
// Tracked on ASYNCcnt (24 ops per call).
// ---------------------------------------------------------------------------
__device__ __forceinline__ void async_copy_wchunk(const unsigned short* __restrict__ wpack,
                                                  unsigned short (*wl)[6144],
                                                  int buf, int ch, int lane) {
#pragma unroll
  for (int mat = 0; mat < 3; ++mat)
#pragma unroll
    for (int n = 0; n < 4; ++n) {
      const unsigned short* src =
          wpack + ((size_t)((mat * 32 + ch) * 4 + n) * 32) * 16 + lane * 16;
      unsigned lds = (unsigned)(uintptr_t)(&wl[buf][(mat * 4 + n) * 512 + lane * 16]);
      unsigned long long ga = (unsigned long long)(uintptr_t)src;
      asm volatile("global_load_async_to_lds_b128 %0, %1, off"
                   :: "v"(lds), "v"(ga) : "memory");
      asm volatile("global_load_async_to_lds_b128 %0, %1, off offset:16"
                   :: "v"(lds), "v"(ga) : "memory");
    }
}

// ---------------------------------------------------------------------------
// Kernel 1: fused QKV projection for one 16-row tile (one wave per block).
// Weights stream through LDS via double-buffered async copies; WMMAs are fed
// from ds_load_b128 so the matrix pipe never waits on global latency.
// q -> row-major bf16 [B*T, H]
// k -> kpack[b][s16][hc][lane][j] = k[s16*16 + lane&15][hc*32 + 16*(lane>>4) + j]
// v -> vhalf[b][s16][n][hcol][j]  = v[s16*16 + j][n*16 + hcol]
// ---------------------------------------------------------------------------
__global__ void __launch_bounds__(32)
qkv_kernel(const float* __restrict__ x, const unsigned short* __restrict__ wpack,
           unsigned short* __restrict__ qbf, unsigned short* __restrict__ kpack,
           unsigned short* __restrict__ vhalf) {
  __shared__ unsigned short wlds[2][6144];   // 2 x 12KB weight chunk buffers
  __shared__ unsigned short tk[16][64];
  __shared__ unsigned short tv[16][64];

  int lane = threadIdx.x;
  int lrow = lane & 15;
  int lhi  = lane >> 4;
  int tile = blockIdx.x;                 // 0..2047
  int b    = tile >> 7;
  int t0   = (tile & 127) << 4;

  const float* xrow = x + ((size_t)(b * TT + t0 + lrow)) * CC;

  f32x8 aq[4], ak[4], av[4];
#pragma unroll
  for (int n = 0; n < 4; ++n) { aq[n] = (f32x8)0.0f; ak[n] = (f32x8)0.0f; av[n] = (f32x8)0.0f; }

  // Prologue: stage chunk 0
  async_copy_wchunk(wpack, wlds, 0, 0, lane);

#pragma unroll 1
  for (int ch = 0; ch < 32; ++ch) {
    int c0  = ch << 5;
    int chn = (ch + 1) & 31;             // last iteration issues a harmless dummy
    async_copy_wchunk(wpack, wlds, chn & 1, chn, lane);

    __builtin_prefetch(xrow + c0 + 256, 0, 1);   // global_prefetch_b8
    // A-fragment from x (fp32 -> bf16): K = j + 16*lhi
    BFrag a;
    float fv[16];
    *(float4*)&fv[0]  = *(const float4*)(xrow + c0 + lhi * 16);
    *(float4*)&fv[4]  = *(const float4*)(xrow + c0 + lhi * 16 + 4);
    *(float4*)&fv[8]  = *(const float4*)(xrow + c0 + lhi * 16 + 8);
    *(float4*)&fv[12] = *(const float4*)(xrow + c0 + lhi * 16 + 12);
#pragma unroll
    for (int j = 0; j < 16; ++j) a.h[j] = f2bf_bits(fv[j]);

    // Only the 24 newest async ops (next chunk) may remain outstanding.
    asm volatile("s_wait_asynccnt 0x18" ::: "memory");

    const unsigned short* bufbase = &wlds[ch & 1][lane * 16];
#pragma unroll
    for (int n = 0; n < 4; ++n) {
      BFrag bq, bk, bv;
      const unsigned short* p0 = bufbase + (0 * 4 + n) * 512;
      const unsigned short* p1 = bufbase + (1 * 4 + n) * 512;
      const unsigned short* p2 = bufbase + (2 * 4 + n) * 512;
      bq.u[0] = *(const uint4*)p0;  bq.u[1] = *(const uint4*)(p0 + 8);
      bk.u[0] = *(const uint4*)p1;  bk.u[1] = *(const uint4*)(p1 + 8);
      bv.u[0] = *(const uint4*)p2;  bv.u[1] = *(const uint4*)(p2 + 8);
      aq[n] = __builtin_amdgcn_wmma_f32_16x16x32_bf16(false, a.f, false, bq.f, (short)0, aq[n], false, false);
      ak[n] = __builtin_amdgcn_wmma_f32_16x16x32_bf16(false, a.f, false, bk.f, (short)0, ak[n], false, false);
      av[n] = __builtin_amdgcn_wmma_f32_16x16x32_bf16(false, a.f, false, bv.f, (short)0, av[n], false, false);
    }
  }

  // Epilogue: q row-major; k,v staged to LDS then packed into fragment layouts.
#pragma unroll
  for (int n = 0; n < 4; ++n) {
#pragma unroll
    for (int r = 0; r < 8; ++r) {
      int row = r + (lhi << 3);
      int h   = n * 16 + lrow;
      qbf[((size_t)(b * TT + t0 + row)) * HH + h] = f2bf_bits(aq[n][r]);
      tk[row][h] = f2bf_bits(ak[n][r]);
      tv[row][h] = f2bf_bits(av[n][r]);
    }
  }
  __syncthreads();

  int s16 = t0 >> 4;
#pragma unroll
  for (int hc = 0; hc < 2; ++hc) {
    const unsigned short* src = &tk[lrow][hc * 32 + (lhi << 4)];
    unsigned short* dst = kpack + (((size_t)(b * NTILE + s16) * 2 + hc) * 32 + lane) * 16;
    *(uint4*)dst       = *(const uint4*)src;
    *(uint4*)(dst + 8) = *(const uint4*)(src + 8);
  }
#pragma unroll
  for (int e0 = 0; e0 < 2; ++e0) {
    int e = lane + e0 * 32;               // 64 entries: (n, hcol)
    int n = e >> 4, hcol = e & 15;
    unsigned int w[8];
#pragma unroll
    for (int p = 0; p < 8; ++p)
      w[p] = (unsigned int)tv[2 * p][n * 16 + hcol] |
             ((unsigned int)tv[2 * p + 1][n * 16 + hcol] << 16);
    unsigned short* dst = vhalf + (((size_t)(b * NTILE + s16) * 4 + n) * 16 + hcol) * 16;
    *(uint4*)dst       = *(uint4*)&w[0];
    *(uint4*)(dst + 8) = *(uint4*)&w[4];
  }
}

// ---------------------------------------------------------------------------
// Kernel 2: flash attention, one wave per 16-query tile, 32 keys per step.
// All k AND v fragment loads are issued in one clause at the top of each
// iteration; V loads drain behind the softmax VALU/trans work.
// ---------------------------------------------------------------------------
__global__ void __launch_bounds__(32)
attn_kernel(const unsigned short* __restrict__ qbf, const unsigned short* __restrict__ kpack,
            const unsigned short* __restrict__ vhalf, float* __restrict__ out) {
  __shared__ unsigned short pls[16][32];

  int lane = threadIdx.x;
  int lrow = lane & 15;
  int lhi  = lane >> 4;
  int tile = blockIdx.x;
  int b    = tile >> 7;
  int t0   = (tile & 127) << 4;

  // q A-fragments (two 32-wide K chunks over H=64)
  BFrag qa[2];
  const unsigned short* qrow = qbf + ((size_t)(b * TT + t0 + lrow)) * HH;
#pragma unroll
  for (int hc = 0; hc < 2; ++hc) {
    qa[hc].u[0] = *(const uint4*)(qrow + hc * 32 + lhi * 16);
    qa[hc].u[1] = *(const uint4*)(qrow + hc * 32 + lhi * 16 + 8);
  }

  f32x8 o[4];
#pragma unroll
  for (int n = 0; n < 4; ++n) o[n] = (f32x8)0.0f;
  float m[8], lsum[8], al[8];
#pragma unroll
  for (int r = 0; r < 8; ++r) { m[r] = -1.0e30f; lsum[r] = 0.0f; }

  const float scale = 0.125f;                  // H^-0.5
#pragma unroll 1
  for (int s0 = 0; s0 < t0 + 16; s0 += 32) {
    int s16 = s0 >> 4;

    // ---- issue ALL fragment loads for this step up front (one clause) ----
    BFrag kb[2][2], vb[4];
#pragma unroll
    for (int n = 0; n < 2; ++n)
#pragma unroll
      for (int hc = 0; hc < 2; ++hc) {
        const unsigned short* kp =
            kpack + (((size_t)(b * NTILE + s16 + n) * 2 + hc) * 32 + lane) * 16;
        kb[n][hc].u[0] = *(const uint4*)kp;
        kb[n][hc].u[1] = *(const uint4*)(kp + 8);
      }
#pragma unroll
    for (int n = 0; n < 4; ++n) {
      const unsigned short* vp =
          vhalf + (((size_t)(b * NTILE + s16 + lhi) * 4 + n) * 16 + lrow) * 16;
      vb[n].u[0] = *(const uint4*)vp;
      vb[n].u[1] = *(const uint4*)(vp + 8);
    }
    // prefetch next step's k/v lines
    __builtin_prefetch(kpack + (((size_t)(b * NTILE + s16 + 2) * 2) * 32 + lane) * 16, 0, 1);
    __builtin_prefetch(vhalf + (((size_t)(b * NTILE + s16 + 2 + lhi) * 4) * 16 + lrow) * 16, 0, 1);

    // S = q . k^T   (two 16x16 N-tiles, K=64 split into 2 chunks)
    f32x8 sacc[2];
    sacc[0] = (f32x8)0.0f; sacc[1] = (f32x8)0.0f;
#pragma unroll
    for (int n = 0; n < 2; ++n)
#pragma unroll
      for (int hc = 0; hc < 2; ++hc)
        sacc[n] = __builtin_amdgcn_wmma_f32_16x16x32_bf16(false, qa[hc].f, false, kb[n][hc].f,
                                                          (short)0, sacc[n], false, false);

    // Online softmax (rows live across 16-lane halves; M = r + 8*lhi, N = lane%16)
#pragma unroll
    for (int r = 0; r < 8; ++r) {
      float v0 = sacc[0][r] * scale;
      float v1 = sacc[1][r] * scale;
      int trow = t0 + r + (lhi << 3);
      if (s0 + lrow      > trow) v0 = -1.0e30f;
      if (s0 + 16 + lrow > trow) v1 = -1.0e30f;
      float mrow = rowmax16(fmaxf(v0, v1));
      float mn   = fmaxf(m[r], mrow);
      float p0   = __expf(v0 - mn);
      float p1   = __expf(v1 - mn);
      float srow = rowsum16(p0 + p1);
      al[r]   = __expf(m[r] - mn);
      lsum[r] = lsum[r] * al[r] + srow;
      m[r]    = mn;
      pls[r + (lhi << 3)][lrow]      = f2bf_bits(p0);
      pls[r + (lhi << 3)][lrow + 16] = f2bf_bits(p1);
    }
#pragma unroll
    for (int n = 0; n < 4; ++n)
#pragma unroll
      for (int r = 0; r < 8; ++r) o[n][r] *= al[r];

    __syncthreads();
    // P C-layout -> A-fragment via LDS
    BFrag pa;
    pa.u[0] = *(const uint4*)&pls[lrow][lhi * 16];
    pa.u[1] = *(const uint4*)&pls[lrow][lhi * 16 + 8];
    __syncthreads();

    // O += P . V  (v fragments already resident)
#pragma unroll
    for (int n = 0; n < 4; ++n)
      o[n] = __builtin_amdgcn_wmma_f32_16x16x32_bf16(false, pa.f, false, vb[n].f,
                                                     (short)0, o[n], false, false);
  }

  // Normalize and store fp32 output
#pragma unroll
  for (int n = 0; n < 4; ++n)
#pragma unroll
    for (int r = 0; r < 8; ++r) {
      int trow = t0 + r + (lhi << 3);
      out[((size_t)(b * TT + trow)) * HH + n * 16 + lrow] = o[n][r] / lsum[r];
    }
}

// ---------------------------------------------------------------------------
extern "C" void kernel_launch(void* const* d_in, const int* in_sizes, int n_in,
                              void* d_out, int out_size, void* d_ws, size_t ws_size,
                              hipStream_t stream) {
  (void)in_sizes; (void)n_in; (void)out_size; (void)ws_size;
  const float* x  = (const float*)d_in[0];
  const float* Wq = (const float*)d_in[1];
  const float* Wk = (const float*)d_in[2];
  const float* Wv = (const float*)d_in[3];
  float* out = (float*)d_out;

  unsigned char* ws = (unsigned char*)d_ws;
  // wpack: 3*32*4*32*16 bf16 = 384 KB ; qbf/kpack/vhalf: 4 MB each
  unsigned short* wpack = (unsigned short*)(ws);
  unsigned short* qbf   = (unsigned short*)(ws + 0x60000);
  unsigned short* kpack = (unsigned short*)(ws + 0x60000 + 0x400000);
  unsigned short* vhalf = (unsigned short*)(ws + 0x60000 + 0x800000);

  wpack_kernel<<<dim3(384),  dim3(32), 0, stream>>>(Wq, Wk, Wv, wpack);
  qkv_kernel <<<dim3(2048), dim3(32), 0, stream>>>(x, wpack, qbf, kpack, vhalf);
  attn_kernel<<<dim3(2048), dim3(32), 0, stream>>>(qbf, kpack, vhalf, out);
}
